// DummyTransformerBlock_35974646071934
// MI455X (gfx1250) — matrix operand order repaired and checked by reading, not compiled, not run on hardware
//
#include <hip/hip_runtime.h>
#include <math.h>

typedef _Float16 half_t;
typedef __attribute__((ext_vector_type(16))) _Float16 v16h;
typedef __attribute__((ext_vector_type(8)))  _Float16 v8h;
typedef __attribute__((ext_vector_type(4)))  _Float16 v4h;
typedef __attribute__((ext_vector_type(8)))  float    v8f;

#define DIM   512
#define HID   2048
#define BATCH 4
#define SEQ   4096
#define NROWS (BATCH*SEQ)

// ---------------------------------------------------------------------------
// LayerNorm over rows of [*, DIM] fp32 -> f16. One wave (32 lanes) per row.
// ---------------------------------------------------------------------------
__global__ __launch_bounds__(256) void ln_cast_kernel(
    const float* __restrict__ X, const float* __restrict__ g,
    const float* __restrict__ b, half_t* __restrict__ Out)
{
  const int lane = threadIdx.x & 31;
  const int wid  = threadIdx.x >> 5;
  const long row = (long)blockIdx.x * 8 + wid;
  const float* xr = X + row * DIM;

  float4 d[4];
  float s = 0.f, ss = 0.f;
#pragma unroll
  for (int i = 0; i < 4; ++i) {
    d[i] = ((const float4*)xr)[i * 32 + lane];
    s  += d[i].x + d[i].y + d[i].z + d[i].w;
    ss += d[i].x*d[i].x + d[i].y*d[i].y + d[i].z*d[i].z + d[i].w*d[i].w;
  }
#pragma unroll
  for (int off = 16; off > 0; off >>= 1) {
    s  += __shfl_xor(s,  off, 32);
    ss += __shfl_xor(ss, off, 32);
  }
  const float mu  = s * (1.0f / DIM);
  const float var = ss * (1.0f / DIM) - mu * mu;
  const float inv = rsqrtf(var + 1e-5f);

#pragma unroll
  for (int i = 0; i < 4; ++i) {
    const int c0 = i * 128 + lane * 4;
    const float4 gg = ((const float4*)g)[i * 32 + lane];
    const float4 bb = ((const float4*)b)[i * 32 + lane];
    v4h o;
    o[0] = (half_t)((d[i].x - mu) * inv * gg.x + bb.x);
    o[1] = (half_t)((d[i].y - mu) * inv * gg.y + bb.y);
    o[2] = (half_t)((d[i].z - mu) * inv * gg.z + bb.z);
    o[3] = (half_t)((d[i].w - mu) * inv * gg.w + bb.w);
    *(v4h*)&Out[row * DIM + c0] = o;
  }
}

// ---------------------------------------------------------------------------
// fp32 [R][C] -> f16 transposed [C][R]   (weights, done once per launch)
// ---------------------------------------------------------------------------
__global__ __launch_bounds__(256) void wcastT_kernel(
    const float* __restrict__ W, half_t* __restrict__ WT, int R, int C)
{
  const int i = blockIdx.x * 256 + threadIdx.x;
  if (i >= R * C) return;
  const int r = i / C, c = i - r * C;
  WT[(long)c * R + r] = (half_t)W[i];
}

// f16 [R][C] -> f16 [C][R] (V^T per batch)
__global__ __launch_bounds__(256) void transposeH_kernel(
    const half_t* __restrict__ V, half_t* __restrict__ VT, int R, int C)
{
  const int i = blockIdx.x * 256 + threadIdx.x;
  if (i >= R * C) return;
  const int r = i / C, c = i - r * C;
  VT[(long)c * R + r] = V[i];
}

// ---------------------------------------------------------------------------
// Row softmax: S fp32 [rows][ncols] -> P f16. One 256-thread block per row.
// ---------------------------------------------------------------------------
__global__ __launch_bounds__(256) void softmax_kernel(
    const float* __restrict__ S, half_t* __restrict__ P, int ncols)
{
  __shared__ float red[8];
  const long row = blockIdx.x;
  const float* sr = S + row * (long)ncols;
  half_t*      pr = P + row * (long)ncols;
  const int lane = threadIdx.x & 31, wid = threadIdx.x >> 5;

  float m = -3.4e38f;
  for (int c = threadIdx.x; c < ncols; c += 256) m = fmaxf(m, sr[c]);
#pragma unroll
  for (int off = 16; off > 0; off >>= 1) m = fmaxf(m, __shfl_xor(m, off, 32));
  if (lane == 0) red[wid] = m;
  __syncthreads();
  if (threadIdx.x == 0) {
    float v = red[0];
    for (int i = 1; i < 8; ++i) v = fmaxf(v, red[i]);
    red[0] = v;
  }
  __syncthreads();
  const float M = red[0];
  __syncthreads();

  float sum = 0.f;
  for (int c = threadIdx.x; c < ncols; c += 256) {
    const float e = expf(sr[c] - M);
    pr[c] = (half_t)e;
    sum += e;
  }
#pragma unroll
  for (int off = 16; off > 0; off >>= 1) sum += __shfl_xor(sum, off, 32);
  if (lane == 0) red[wid] = sum;
  __syncthreads();
  if (threadIdx.x == 0) {
    float v = 0.f;
    for (int i = 0; i < 8; ++i) v += red[i];
    red[0] = v;
  }
  __syncthreads();
  const float inv = 1.0f / red[0];
  for (int c = threadIdx.x; c < ncols; c += 256)
    pr[c] = (half_t)((float)pr[c] * inv);
}

// ---------------------------------------------------------------------------
// WMMA GEMM (NT):  C[M][Nn] = A[M][K](f16) * BT[Nn][K](f16)^T  + epilogue
// Block tile 128x128, BK=32, 256 threads (8 waves), wave tile 64x32 (4x2 WMMA)
// Global->LDS staging uses CDNA5 async-LDS loads (ASYNCcnt), triple-buffered
// so the steady-state loop needs a single workgroup barrier per K-step.
// ---------------------------------------------------------------------------
enum { EP_F16_BIAS = 0, EP_F16_GELU = 1, EP_F16_PLAIN = 2,
       EP_F32_SCALE = 3, EP_F32_RES = 4 };

#define LDSW 40   // padded row stride in halfs (80B: conflict-free, 16B aligned)

template<int MODE>
__global__ __launch_bounds__(256) void gemm_nt_kernel(
    const half_t* __restrict__ A, const half_t* __restrict__ BT,
    const float* __restrict__ bias, const float* __restrict__ residual,
    void* __restrict__ Cout, int M, int Nn, int K, float scale)
{
  __shared__ __align__(16) half_t As[3][128 * LDSW];
  __shared__ __align__(16) half_t Bs[3][128 * LDSW];

  const int tid  = threadIdx.x;
  const int lane = tid & 31;
  const int wid  = tid >> 5;
  const int l16  = lane & 15;
  const int hh   = lane >> 4;              // half-wave select
  const long bm  = (long)blockIdx.y * 128;
  const long bn  = (long)blockIdx.x * 128;
  const int  wm  = (wid >> 2) * 64;        // wave M origin in block
  const int  wn  = (wid & 3)  * 32;        // wave N origin in block

  // staging: each thread moves 2 rows x 8 halfs (16B) for A and for B
  const int  sr = tid >> 2;                // 0..63
  const int  sc = (tid & 3) * 8;           // 0,8,16,24
  const half_t* Ab = A  + (bm + sr) * (long)K + sc;
  const half_t* Bb = BT + (bn + sr) * (long)K + sc;
  const long rstep = 64L * K;

  // LDS byte addresses (flat shared ptr keeps the LDS offset in addr[31:0])
  const unsigned ldsA = (unsigned)(size_t)&As[0][sr * LDSW + sc];
  const unsigned ldsB = (unsigned)(size_t)&Bs[0][sr * LDSW + sc];
  const unsigned rowHalf  = 64u * LDSW * 2u;    // rows 64..127
  const unsigned bufBytes = 128u * LDSW * 2u;   // one buffer

  auto issue_tile = [&](int k0, int buf) {
    const unsigned oa = ldsA + (unsigned)buf * bufBytes;
    const unsigned ob = ldsB + (unsigned)buf * bufBytes;
    const unsigned long long ga0 = (unsigned long long)(Ab + k0);
    const unsigned long long ga1 = (unsigned long long)(Ab + rstep + k0);
    const unsigned long long gb0 = (unsigned long long)(Bb + k0);
    const unsigned long long gb1 = (unsigned long long)(Bb + rstep + k0);
    asm volatile("global_load_async_to_lds_b128 %0, %1, off"
                 :: "v"(oa),           "v"(ga0) : "memory");
    asm volatile("global_load_async_to_lds_b128 %0, %1, off"
                 :: "v"(oa + rowHalf), "v"(ga1) : "memory");
    asm volatile("global_load_async_to_lds_b128 %0, %1, off"
                 :: "v"(ob),           "v"(gb0) : "memory");
    asm volatile("global_load_async_to_lds_b128 %0, %1, off"
                 :: "v"(ob + rowHalf), "v"(gb1) : "memory");
  };

  v8f acc[4][2] = {};

  const int nk = K / 32;
  issue_tile(0, 0);
  if (nk > 1) issue_tile(32, 1);

  for (int kt = 0; kt < nk; ++kt) {
    const int buf = kt % 3;
    // Own 4 async loads for tile kt are the oldest outstanding: with tile kt+1
    // (4 more) possibly in flight, ASYNCcnt<=4 proves tile kt landed in LDS.
    if (kt + 1 < nk) asm volatile("s_wait_asynccnt 0x4" ::: "memory");
    else             asm volatile("s_wait_asynccnt 0x0" ::: "memory");
    // One barrier: publishes tile kt from all waves AND proves buffer
    // (kt+2)%3 == (kt-1)%3 is no longer being read -> safe to refill.
    __syncthreads();
    if (kt + 2 < nk) issue_tile((kt + 2) * 32, (kt + 2) % 3);

    const half_t* __restrict__ Ash = As[buf];
    const half_t* __restrict__ Bsh = Bs[buf];

    // Fragments per CDNA5 WMMA VGPR layouts (ISA 7.12.2):
    //  A lane<16: K{0..7,16..23}; lane>=16: K{8..15,24..31}
    //  B lane<16: K{0..7,16..23}; lane>=16: K{8..15,24..31}
    union { v16h v; v8h h[2]; } a[4], bf[2];
#pragma unroll
    for (int i = 0; i < 4; ++i) {
      const int r = wm + i * 16 + l16;
      a[i].h[0] = *(const v8h*)&Ash[r * LDSW + 8 * hh];
      a[i].h[1] = *(const v8h*)&Ash[r * LDSW + 16 + 8 * hh];
    }
#pragma unroll
    for (int j = 0; j < 2; ++j) {
      const int n = wn + j * 16 + l16;
      bf[j].h[0] = *(const v8h*)&Bsh[n * LDSW + 8 * hh];
      bf[j].h[1] = *(const v8h*)&Bsh[n * LDSW + 16 + 8 * hh];
    }
#pragma unroll
    for (int i = 0; i < 4; ++i)
#pragma unroll
      for (int j = 0; j < 2; ++j)
        acc[i][j] = __builtin_amdgcn_wmma_f32_16x16x32_f16(
            false, a[i].v, false, bf[j].v, (short)0, acc[i][j], false, false);
  }

  // Epilogue. C 16x16 layout: VGPR r, lane -> row = r + 8*(lane/16), col = lane%16
#pragma unroll
  for (int i = 0; i < 4; ++i) {
#pragma unroll
    for (int j = 0; j < 2; ++j) {
      const long col  = bn + wn + j * 16 + l16;
      const float bval = (MODE == EP_F16_BIAS || MODE == EP_F16_GELU ||
                          MODE == EP_F32_RES) ? bias[col] : 0.f;
#pragma unroll
      for (int r = 0; r < 8; ++r) {
        const long row = bm + wm + i * 16 + r + 8 * hh;
        const long idx = row * (long)Nn + col;
        const float v  = acc[i][j][r];
        if (MODE == EP_F16_BIAS) {
          ((half_t*)Cout)[idx] = (half_t)(v + bval);
        } else if (MODE == EP_F16_GELU) {
          float t = v + bval;
          t = 0.5f * t * (1.0f + erff(t * 0.7071067811865475f));
          ((half_t*)Cout)[idx] = (half_t)t;
        } else if (MODE == EP_F16_PLAIN) {
          ((half_t*)Cout)[idx] = (half_t)v;
        } else if (MODE == EP_F32_SCALE) {
          ((float*)Cout)[idx] = v * scale;
        } else { // EP_F32_RES
          ((float*)Cout)[idx] = v + bval + residual[idx];
        }
      }
    }
  }
}

// ---------------------------------------------------------------------------
extern "C" void kernel_launch(void* const* d_in, const int* in_sizes, int n_in,
                              void* d_out, int out_size, void* d_ws, size_t ws_size,
                              hipStream_t stream)
{
  const float* x    = (const float*)d_in[0];
  const float* ln1g = (const float*)d_in[1];
  const float* ln1b = (const float*)d_in[2];
  const float* Wq   = (const float*)d_in[3];
  const float* bq   = (const float*)d_in[4];
  const float* Wk   = (const float*)d_in[5];
  const float* bk   = (const float*)d_in[6];
  const float* Wv   = (const float*)d_in[7];
  const float* bv   = (const float*)d_in[8];
  const float* Wo   = (const float*)d_in[9];
  const float* bo   = (const float*)d_in[10];
  const float* ln2g = (const float*)d_in[11];
  const float* ln2b = (const float*)d_in[12];
  const float* W1   = (const float*)d_in[13];
  const float* b1   = (const float*)d_in[14];
  const float* W2   = (const float*)d_in[15];
  const float* b2   = (const float*)d_in[16];

  // ---- workspace carve (S32 aliases u16; h16 doubles as h2) ----
  char* p = (char*)d_ws;
  auto carve = [&](size_t bytes) {
    char* r = p; p += (bytes + 255) & ~(size_t)255; return r;
  };
  half_t* WqT = (half_t*)carve((size_t)DIM * DIM * 2);
  half_t* WkT = (half_t*)carve((size_t)DIM * DIM * 2);
  half_t* WvT = (half_t*)carve((size_t)DIM * DIM * 2);
  half_t* WoT = (half_t*)carve((size_t)DIM * DIM * 2);
  half_t* W1T = (half_t*)carve((size_t)DIM * HID * 2);
  half_t* W2T = (half_t*)carve((size_t)HID * DIM * 2);
  half_t* h16 = (half_t*)carve((size_t)NROWS * DIM * 2);   // ln1 out; reused for ln2
  half_t* q16 = (half_t*)carve((size_t)NROWS * DIM * 2);
  half_t* k16 = (half_t*)carve((size_t)NROWS * DIM * 2);
  half_t* v16 = (half_t*)carve((size_t)NROWS * DIM * 2);
  half_t* o16 = (half_t*)carve((size_t)NROWS * DIM * 2);
  float*  x1  = (float*) carve((size_t)NROWS * DIM * 4);
  half_t* vT  = (half_t*)carve((size_t)DIM * SEQ * 2);
  half_t* P16 = (half_t*)carve((size_t)SEQ * SEQ * 2);
  float*  S32 = (float*) carve((size_t)SEQ * SEQ * 4);     // aliases u16 in time
  half_t* u16 = (half_t*)S32;                              // MLP hidden (after attn)
  if ((size_t)(p - (char*)d_ws) > ws_size) return;         // ws too small: no-op

  // ---- weight cast+transpose (f16, [N][K]) ----
  wcastT_kernel<<<(DIM * DIM + 255) / 256, 256, 0, stream>>>(Wq, WqT, DIM, DIM);
  wcastT_kernel<<<(DIM * DIM + 255) / 256, 256, 0, stream>>>(Wk, WkT, DIM, DIM);
  wcastT_kernel<<<(DIM * DIM + 255) / 256, 256, 0, stream>>>(Wv, WvT, DIM, DIM);
  wcastT_kernel<<<(DIM * DIM + 255) / 256, 256, 0, stream>>>(Wo, WoT, DIM, DIM);
  wcastT_kernel<<<(DIM * HID + 255) / 256, 256, 0, stream>>>(W1, W1T, DIM, HID);
  wcastT_kernel<<<(HID * DIM + 255) / 256, 256, 0, stream>>>(W2, W2T, HID, DIM);

  // ---- LN1 -> h ----
  ln_cast_kernel<<<NROWS / 8, 256, 0, stream>>>(x, ln1g, ln1b, h16);

  // ---- QKV projections ----
  const dim3 gDD(DIM / 128, NROWS / 128);
  gemm_nt_kernel<EP_F16_BIAS><<<gDD, 256, 0, stream>>>(h16, WqT, bq, nullptr, q16, NROWS, DIM, DIM, 1.f);
  gemm_nt_kernel<EP_F16_BIAS><<<gDD, 256, 0, stream>>>(h16, WkT, bk, nullptr, k16, NROWS, DIM, DIM, 1.f);
  gemm_nt_kernel<EP_F16_BIAS><<<gDD, 256, 0, stream>>>(h16, WvT, bv, nullptr, v16, NROWS, DIM, DIM, 1.f);

  // ---- attention, per batch (materialized scores; HBM-cheap at 23 TB/s) ----
  const float scal = 0.04419417382415922f;  // 1/sqrt(512)
  for (int bch = 0; bch < BATCH; ++bch) {
    const half_t* qb = q16 + (long)bch * SEQ * DIM;
    const half_t* kb = k16 + (long)bch * SEQ * DIM;
    const half_t* vb = v16 + (long)bch * SEQ * DIM;
    half_t*       ob = o16 + (long)bch * SEQ * DIM;
    // S = scale * q @ k^T   (k is [S][D] == BT layout already)
    gemm_nt_kernel<EP_F32_SCALE><<<dim3(SEQ / 128, SEQ / 128), 256, 0, stream>>>(
        qb, kb, nullptr, nullptr, S32, SEQ, SEQ, DIM, scal);
    softmax_kernel<<<SEQ, 256, 0, stream>>>(S32, P16, SEQ);
    transposeH_kernel<<<(SEQ * DIM + 255) / 256, 256, 0, stream>>>(vb, vT, SEQ, DIM);
    // o = P @ V  (BT = V^T [D][S])
    gemm_nt_kernel<EP_F16_PLAIN><<<dim3(DIM / 128, SEQ / 128), 256, 0, stream>>>(
        P16, vT, nullptr, nullptr, ob, SEQ, DIM, SEQ, 1.f);
  }

  // ---- x1 = x + o @ Wo + bo ----
  gemm_nt_kernel<EP_F32_RES><<<gDD, 256, 0, stream>>>(o16, WoT, bo, x, x1, NROWS, DIM, DIM, 1.f);

  // ---- LN2 -> h2 (reuse h16) ----
  ln_cast_kernel<<<NROWS / 8, 256, 0, stream>>>(x1, ln2g, ln2b, h16);

  // ---- MLP: u = gelu(h2 @ W1 + b1); out = x1 + u @ W2 + b2 ----
  gemm_nt_kernel<EP_F16_GELU><<<dim3(HID / 128, NROWS / 128), 256, 0, stream>>>(
      h16, W1T, b1, nullptr, u16, NROWS, HID, DIM, 1.f);
  gemm_nt_kernel<EP_F32_RES><<<gDD, 256, 0, stream>>>(
      u16, W2T, b2, x1, (float*)d_out, NROWS, DIM, HID, 1.f);
}
